// Mamba2_time_series_12395275616351
// MI455X (gfx1250) — matrix operand order repaired and checked by reading, not compiled, not run on hardware
//
#include <hip/hip_runtime.h>
#include <hip/hip_bf16.h>
#include <stdint.h>

// ---------------- model dims ----------------
#define D_MODEL   768
#define D_STATE   128
#define HEADDIM   128
#define D_INNER   1536
#define NHEADS    12
#define D_CONV    4
#define CONV_DIM  1792            // D_INNER + 2*D_STATE
#define D_IN      3340            // 2*D_INNER + 2*D_STATE + NHEADS
#define NLAYERS   4
#define EPS_F     1e-5f
#define CHUNKQ    125
#define NB        8
#define L_IN      2500
#define IN_CH     12
#define NUM_CLASSES 27
#define HID       1024
#define LSEQ      625             // sequence length after conv front-end
#define NCH       5               // LSEQ / CHUNKQ
#define MROWS     5000            // NB * LSEQ
#define MPAD      5056            // padded to /64 (wave macro-tile M)
#define NPAD_IN   3360            // D_IN padded to /32 (wave macro-tile N)

typedef __attribute__((ext_vector_type(16))) __bf16 v16bf;
typedef __attribute__((ext_vector_type(8)))  float  v8f;

__device__ __forceinline__ float gelu_exact(float x) {
  return 0.5f * x * (1.0f + erff(x * 0.70710678118654752f));
}
__device__ __forceinline__ float silu_f(float x) { return x / (1.0f + expf(-x)); }
__device__ __forceinline__ float softplus_f(float x) {
  return (x > 20.f) ? x : log1pf(expf(x));
}

// ---------------- conv front-end: conv1d + bias + GELU + BN ----------------
__global__ void conv_front(const float* __restrict__ in, const float* __restrict__ w,
                           const float* __restrict__ bias, const float* __restrict__ bn,
                           float* __restrict__ out, int Bn, int Cin, int Lin,
                           int Cout, int Lout, int Kw, int stride, int pad,
                           int transpose_out) {
  int idx = blockIdx.x * blockDim.x + threadIdx.x;
  int total = Bn * Cout * Lout;
  if (idx >= total) return;
  int lo = idx % Lout;
  int co = (idx / Lout) % Cout;
  int b  = idx / (Lout * Cout);
  float s = bias[co];
  int l0 = lo * stride - pad;
  for (int ci = 0; ci < Cin; ++ci) {
    const float* ir = in + ((size_t)b * Cin + ci) * Lin;
    const float* wr = w + ((size_t)co * Cin + ci) * Kw;
    for (int j = 0; j < Kw; ++j) {
      int li = l0 + j;
      if (li >= 0 && li < Lin) s += wr[j] * ir[li];
    }
  }
  float g  = gelu_exact(s);
  float gg = bn[co], bb = bn[Cout + co], mm = bn[2 * Cout + co], vv = bn[3 * Cout + co];
  float scale = gg * rsqrtf(vv + EPS_F);
  float val = g * scale + (bb - mm * scale);
  if (transpose_out) out[((size_t)b * Lout + lo) * Cout + co] = val;
  else               out[((size_t)b * Cout + co) * Lout + lo] = val;
}

// ---------------- LayerNorm(768) -> bf16 activation matrix (padded rows zeroed) ----
__global__ void ln_to_bf16(const float* __restrict__ hs, const float* __restrict__ g,
                           const float* __restrict__ bvec, __bf16* __restrict__ A,
                           int Mreal, int D) {
  int row = blockIdx.x;
  int tid = threadIdx.x;
  if (row >= Mreal) {
    for (int d = tid; d < D; d += blockDim.x) A[(size_t)row * D + d] = (__bf16)0.f;
    return;
  }
  __shared__ float s1[256], s2[256];
  const float* hr = hs + (size_t)row * D;
  float sum = 0.f, sq = 0.f;
  for (int d = tid; d < D; d += 256) { float v = hr[d]; sum += v; sq += v * v; }
  s1[tid] = sum; s2[tid] = sq; __syncthreads();
  for (int o = 128; o > 0; o >>= 1) {
    if (tid < o) { s1[tid] += s1[tid + o]; s2[tid] += s2[tid + o]; }
    __syncthreads();
  }
  float mu = s1[0] / (float)D;
  float var = s2[0] / (float)D - mu * mu;
  float rs = rsqrtf(var + EPS_F);
  for (int d = tid; d < D; d += 256) {
    float v = (hr[d] - mu) * rs * g[d] + bvec[d];
    A[(size_t)row * D + d] = (__bf16)v;
  }
}

// ---------------- weights f32 -> bf16 with N padding ----------------
__global__ void conv_w_bf16(const float* __restrict__ W, __bf16* __restrict__ Wb,
                            int Nreal, int Npad, int K) {
  int idx = blockIdx.x * blockDim.x + threadIdx.x;
  int total = Npad * K;
  if (idx >= total) return;
  int n = idx / K, k = idx % K;
  Wb[idx] = (n < Nreal) ? (__bf16)W[(size_t)n * K + k] : (__bf16)0.f;
}

// ---------------- BF16 WMMA GEMM: C[M,N] = A[M,K] * W[N,K]^T -------------------
// One wave computes a 64x32 macro-tile (4x2 grid of 16x16 WMMA tiles).
// Double-buffered fragment pipeline with STATIC indexing only (no runtime
// array indices), so all fragments stay in VGPRs (no scratch/LDS demotion).
// K must be a multiple of 32 (768 and 1536 here).
__global__ void __launch_bounds__(256)
gemm_bf16_wmma(const __bf16* __restrict__ A, const __bf16* __restrict__ Bw,
               float* __restrict__ C, int Mt64, int Nt32, int K,
               int lda, int ldb, int ldc) {
  int wave = threadIdx.x >> 5;
  int lane = threadIdx.x & 31;
  int tile = blockIdx.x * 8 + wave;
  if (tile >= Mt64 * Nt32) return;           // wave-uniform: EXEC stays all-ones
  int tm = tile / Nt32, tn = tile % Nt32;
  int hi = lane >> 4, lo = lane & 15;

  // Per-lane base pointers (hi offset folded in).
  const __bf16* Ar[4];
  const __bf16* Br[2];
#pragma unroll
  for (int mi = 0; mi < 4; ++mi)
    Ar[mi] = A + (size_t)(tm * 64 + mi * 16 + lo) * lda + hi * 8;
#pragma unroll
  for (int ni = 0; ni < 2; ++ni)
    Br[ni] = Bw + (size_t)(tn * 32 + ni * 16 + lo) * ldb + hi * 8;

  union Frag16 { v16bf v; uint4 q[2]; };
  // A 16x32 fragment: 16B runs at k and k+16 (relative to lane base)
#define LOADA(f, ptr, kk) { const uint4* _ap = reinterpret_cast<const uint4*>((ptr) + (kk)); \
                            (f).q[0] = _ap[0]; (f).q[1] = _ap[2]; }
  // B 32x16 fragment: 16B runs at k and k+16 (relative to lane base)
#define LOADB(f, ptr, kk) { const uint4* _bp = reinterpret_cast<const uint4*>((ptr) + (kk)); \
                            (f).q[0] = _bp[0]; (f).q[1] = _bp[2]; }

  v8f acc[4][2];
#pragma unroll
  for (int mi = 0; mi < 4; ++mi)
#pragma unroll
    for (int ni = 0; ni < 2; ++ni) { v8f z = {}; acc[mi][ni] = z; }

  Frag16 a0[4], b0[2], a1[4], b1[2];
#pragma unroll
  for (int mi = 0; mi < 4; ++mi) LOADA(a0[mi], Ar[mi], 0);
#pragma unroll
  for (int ni = 0; ni < 2; ++ni) LOADB(b0[ni], Br[ni], 0);

  int k0 = 0;
  while (true) {
    int k1 = k0 + 32;
    bool more1 = (k1 < K);                   // scalar-uniform
    if (more1) {
#pragma unroll
      for (int mi = 0; mi < 4; ++mi) LOADA(a1[mi], Ar[mi], k1);
#pragma unroll
      for (int ni = 0; ni < 2; ++ni) LOADB(b1[ni], Br[ni], k1);
    }
#pragma unroll
    for (int mi = 0; mi < 4; ++mi)
#pragma unroll
      for (int ni = 0; ni < 2; ++ni)
        acc[mi][ni] = __builtin_amdgcn_wmma_f32_16x16x32_bf16(
            false, a0[mi].v, false, b0[ni].v, (short)0, acc[mi][ni], false, false);
    if (!more1) break;

    int k2 = k0 + 64;
    bool more2 = (k2 < K);
    if (more2) {
#pragma unroll
      for (int mi = 0; mi < 4; ++mi) LOADA(a0[mi], Ar[mi], k2);
#pragma unroll
      for (int ni = 0; ni < 2; ++ni) LOADB(b0[ni], Br[ni], k2);
    }
#pragma unroll
    for (int mi = 0; mi < 4; ++mi)
#pragma unroll
      for (int ni = 0; ni < 2; ++ni)
        acc[mi][ni] = __builtin_amdgcn_wmma_f32_16x16x32_bf16(
            false, a1[mi].v, false, b1[ni].v, (short)0, acc[mi][ni], false, false);
    if (!more2) break;
    k0 = k2;
  }
#undef LOADA
#undef LOADB

  // D layout: lane col n = (lane&15); VGPR r -> m = r + hi*8
  int ncol0 = tn * 32 + lo;
#pragma unroll
  for (int mi = 0; mi < 4; ++mi) {
    int mbase = tm * 64 + mi * 16 + hi * 8;
#pragma unroll
    for (int ni = 0; ni < 2; ++ni) {
      int ncol = ncol0 + ni * 16;
#pragma unroll
      for (int r = 0; r < 8; ++r)
        C[(size_t)(mbase + r) * ldc + ncol] = acc[mi][ni][r];
    }
  }
}

// ---------------- depthwise causal conv (k=4) + SiLU over xBC slice ------------
__global__ void dwconv_silu(const float* __restrict__ zx, const float* __restrict__ cw,
                            const float* __restrict__ cb, float* __restrict__ xBC,
                            int Bn, int L, int ldz) {
  int idx = blockIdx.x * blockDim.x + threadIdx.x;
  int total = Bn * L * CONV_DIM;
  if (idx >= total) return;
  int c = idx % CONV_DIM;
  int l = (idx / CONV_DIM) % L;
  int b = idx / (CONV_DIM * L);
  float s = cb[c];
#pragma unroll
  for (int j = 0; j < D_CONV; ++j) {
    int li = l - (D_CONV - 1) + j;
    if (li >= 0)
      s += cw[c * D_CONV + j] * zx[((size_t)(b * L + li)) * ldz + D_INNER + c];
  }
  xBC[idx] = silu_f(s);
}

// ---------------- dt = softplus(dt_raw + bias) ----------------
__global__ void dt_prep(const float* __restrict__ zx, const float* __restrict__ dt_bias,
                        float* __restrict__ dtb, int ldz) {
  int idx = blockIdx.x * blockDim.x + threadIdx.x;
  int total = NB * LSEQ * NHEADS;
  if (idx >= total) return;
  int h = idx % NHEADS;
  int row = idx / NHEADS;
  dtb[idx] = softplus_f(zx[(size_t)row * ldz + D_INNER + CONV_DIM + h] + dt_bias[h]);
}

// ---------------- per-(b,c,h) cumsum of a = dt * (-exp(A_log)) ----------------
__global__ void cumsum_a(const float* __restrict__ dtb, const float* __restrict__ A_log,
                         float* __restrict__ Acs) {
  int tid = blockIdx.x * blockDim.x + threadIdx.x;
  if (tid >= NB * NCH * NHEADS) return;
  int h = tid % NHEADS;
  int c = (tid / NHEADS) % NCH;
  int b = tid / (NHEADS * NCH);
  float negA = -expf(A_log[h]);
  float run = 0.f;
  int base = (b * LSEQ + c * CHUNKQ) * NHEADS + h;
  for (int q = 0; q < CHUNKQ; ++q) {
    run += dtb[base + q * NHEADS] * negA;
    Acs[tid * CHUNKQ + q] = run;
  }
}

// ---------------- scores[b,c,l,s] = sum_n C[l,n] * B[s,n] ----------------
__global__ void scores_k(const float* __restrict__ xBC, float* __restrict__ scores) {
  int idx = blockIdx.x * blockDim.x + threadIdx.x;
  int total = NB * NCH * CHUNKQ * CHUNKQ;
  if (idx >= total) return;
  int s = idx % CHUNKQ;
  int l = (idx / CHUNKQ) % CHUNKQ;
  int bc = idx / (CHUNKQ * CHUNKQ);
  int b = bc / NCH, c = bc % NCH;
  const float* Cr = xBC + ((size_t)(b * LSEQ + c * CHUNKQ + l)) * CONV_DIM + D_INNER + D_STATE;
  const float* Br = xBC + ((size_t)(b * LSEQ + c * CHUNKQ + s)) * CONV_DIM + D_INNER;
  float acc = 0.f;
  for (int n = 0; n < D_STATE; ++n) acc += Cr[n] * Br[n];
  scores[idx] = acc;
}

// ---------------- Y_diag: block per (b,c,h); X tile in LDS ----------------
__global__ void __launch_bounds__(128)
ydiag_k(const float* __restrict__ scores, const float* __restrict__ Acs,
        const float* __restrict__ xBC, const float* __restrict__ dtb,
        float* __restrict__ ybuf) {
  __shared__ float Xs[CHUNKQ * HEADDIM];   // 64000 B
  __shared__ float AcsS[CHUNKQ];
  int bch = blockIdx.x;                    // (b*NCH + c)*NHEADS + h
  int h = bch % NHEADS;
  int c = (bch / NHEADS) % NCH;
  int b = bch / (NHEADS * NCH);
  int p = threadIdx.x;                     // 0..127
  int rowbase = b * LSEQ + c * CHUNKQ;
  for (int s = 0; s < CHUNKQ; ++s) {
    float dt = dtb[(size_t)(rowbase + s) * NHEADS + h];
    Xs[s * HEADDIM + p] = xBC[(size_t)(rowbase + s) * CONV_DIM + h * HEADDIM + p] * dt;
  }
  if (p < CHUNKQ) AcsS[p] = Acs[(size_t)bch * CHUNKQ + p];
  __syncthreads();
  const float* sc = scores + ((size_t)(b * NCH + c)) * CHUNKQ * CHUNKQ;
  for (int l = 0; l < CHUNKQ; ++l) {
    float al = AcsS[l];
    float acc = 0.f;
    for (int s = 0; s <= l; ++s)
      acc += sc[l * CHUNKQ + s] * expf(al - AcsS[s]) * Xs[s * HEADDIM + p];
    ybuf[((size_t)(rowbase + l) * NHEADS + h) * HEADDIM + p] = acc;
  }
}

// ---------------- chunk states[b,c,h,p,n] = sum_l X[l,p]*B[l,n]*decay[l] --------
__global__ void states_k(const float* __restrict__ xBC, const float* __restrict__ dtb,
                         const float* __restrict__ Acs, float* __restrict__ states) {
  int idx = blockIdx.x * blockDim.x + threadIdx.x;
  int total = NB * NCH * NHEADS * HEADDIM * D_STATE;
  if (idx >= total) return;
  int n = idx % D_STATE;
  int p = (idx / D_STATE) % HEADDIM;
  int bch = idx / (D_STATE * HEADDIM);
  int h = bch % NHEADS;
  int c = (bch / NHEADS) % NCH;
  int b = bch / (NHEADS * NCH);
  int rowbase = b * LSEQ + c * CHUNKQ;
  float alast = Acs[(size_t)bch * CHUNKQ + (CHUNKQ - 1)];
  float acc = 0.f;
  for (int l = 0; l < CHUNKQ; ++l) {
    const float* xr = xBC + (size_t)(rowbase + l) * CONV_DIM;
    float ds = expf(alast - Acs[(size_t)bch * CHUNKQ + l]);
    float x = xr[h * HEADDIM + p] * dtb[(size_t)(rowbase + l) * NHEADS + h];
    acc += x * xr[D_INNER + n] * ds;
  }
  states[idx] = acc;
}

// ---------------- in-place chunk scan: states[c] becomes s_in[c] ----------------
__global__ void scan_k(float* __restrict__ states, const float* __restrict__ Acs) {
  int idx = blockIdx.x * blockDim.x + threadIdx.x;
  int total = NB * NHEADS * HEADDIM * D_STATE;
  if (idx >= total) return;
  int n = idx % D_STATE;
  int p = (idx / D_STATE) % HEADDIM;
  int h = (idx / (D_STATE * HEADDIM)) % NHEADS;
  int b = idx / (D_STATE * HEADDIM * NHEADS);
  float s = 0.f;
  for (int c = 0; c < NCH; ++c) {
    int bch = (b * NCH + c) * NHEADS + h;
    float cd = expf(Acs[(size_t)bch * CHUNKQ + (CHUNKQ - 1)]);
    size_t sidx = ((size_t)bch * HEADDIM + p) * D_STATE + n;
    float tmp = states[sidx];
    states[sidx] = s;                     // s_in[c] = carry before this chunk
    s = s * cd + tmp;
  }
}

// ---------------- Y_off + Y_diag + D*xh -> ybuf ----------------
__global__ void yoff_k(const float* __restrict__ xBC, const float* __restrict__ dtb,
                       const float* __restrict__ Acs, const float* __restrict__ s_in,
                       const float* __restrict__ Dp, float* __restrict__ ybuf) {
  int idx = blockIdx.x * blockDim.x + threadIdx.x;
  int total = NB * LSEQ * NHEADS * HEADDIM;
  if (idx >= total) return;
  int p = idx % HEADDIM;
  int h = (idx / HEADDIM) % NHEADS;
  int lg = idx / (HEADDIM * NHEADS);      // global row b*LSEQ + l
  int b = lg / LSEQ;
  int ll = lg % LSEQ;
  int c = ll / CHUNKQ;
  int l = ll % CHUNKQ;
  int bch = (b * NCH + c) * NHEADS + h;
  float eA = expf(Acs[(size_t)bch * CHUNKQ + l]);
  const float* Cr = xBC + (size_t)lg * CONV_DIM + D_INNER + D_STATE;
  const float* sr = s_in + ((size_t)bch * HEADDIM + p) * D_STATE;
  float acc = 0.f;
  for (int n = 0; n < D_STATE; ++n) acc += Cr[n] * sr[n];
  float xh = xBC[(size_t)lg * CONV_DIM + h * HEADDIM + p];
  ybuf[idx] = ybuf[idx] + eA * acc + Dp[h] * xh;
}

// ---------------- gate by silu(z), RMSNorm, scale, -> bf16 (padded) ------------
__global__ void gate_rms_bf16(const float* __restrict__ ybuf, const float* __restrict__ zx,
                              const float* __restrict__ rms_w, __bf16* __restrict__ A2,
                              int Mreal, int ldz) {
  int row = blockIdx.x;
  int tid = threadIdx.x;
  if (row >= Mreal) {
    for (int d = tid; d < D_INNER; d += 256) A2[(size_t)row * D_INNER + d] = (__bf16)0.f;
    return;
  }
  __shared__ float red[256];
  float yg[6];
  float sq = 0.f;
#pragma unroll
  for (int k = 0; k < 6; ++k) {
    int d = tid + k * 256;
    float v = ybuf[(size_t)row * D_INNER + d] * silu_f(zx[(size_t)row * ldz + d]);
    yg[k] = v; sq += v * v;
  }
  red[tid] = sq; __syncthreads();
  for (int o = 128; o > 0; o >>= 1) {
    if (tid < o) red[tid] += red[tid + o];
    __syncthreads();
  }
  float rs = rsqrtf(red[0] / (float)D_INNER + EPS_F);
#pragma unroll
  for (int k = 0; k < 6; ++k) {
    int d = tid + k * 256;
    A2[(size_t)row * D_INNER + d] = (__bf16)(yg[k] * rs * rms_w[d]);
  }
}

// ---------------- residual add ----------------
__global__ void add_res(float* __restrict__ hs, const float* __restrict__ gout) {
  int idx = blockIdx.x * blockDim.x + threadIdx.x;
  if (idx >= MROWS * D_MODEL) return;
  hs[idx] += gout[idx];                    // gout ldc == D_MODEL, rows contiguous
}

// ---------------- masked mean pool ----------------
__global__ void pool_k(const float* __restrict__ hs, const float* __restrict__ mask,
                       float* __restrict__ rep) {
  __shared__ float mdsh[LSEQ];
  __shared__ float red[256];
  int b = blockIdx.x, tid = threadIdx.x;
  float part = 0.f;
  for (int l = tid; l < LSEQ; l += 256) {
    const float* mr = mask + (size_t)b * L_IN + l * 4;
    float m4 = fmaxf(fmaxf(mr[0], mr[1]), fmaxf(mr[2], mr[3]));
    mdsh[l] = m4; part += m4;
  }
  red[tid] = part; __syncthreads();
  for (int o = 128; o > 0; o >>= 1) {
    if (tid < o) red[tid] += red[tid + o];
    __syncthreads();
  }
  float denom = fmaxf(red[0], 1e-9f);
  for (int d = tid; d < D_MODEL; d += 256) {
    float acc = 0.f;
    for (int l = 0; l < LSEQ; ++l)
      acc += hs[((size_t)(b * LSEQ + l)) * D_MODEL + d] * mdsh[l];
    rep[b * D_MODEL + d] = acc / denom;
  }
}

// ---------------- classifier ----------------
__global__ void cls1_k(const float* __restrict__ rep, const float* __restrict__ w1,
                       const float* __restrict__ b1, const float* __restrict__ bn,
                       float* __restrict__ h1c) {
  int idx = blockIdx.x * blockDim.x + threadIdx.x;
  if (idx >= NB * HID) return;
  int j = idx % HID, b = idx / HID;
  const float* rr = rep + (size_t)b * D_MODEL;
  const float* wr = w1 + (size_t)j * D_MODEL;
  float s = b1[j];
  for (int k = 0; k < D_MODEL; ++k) s += rr[k] * wr[k];
  float gg = bn[j], bb = bn[HID + j], mm = bn[2 * HID + j], vv = bn[3 * HID + j];
  float scale = gg * rsqrtf(vv + EPS_F);
  h1c[idx] = fmaxf(s * scale + (bb - mm * scale), 0.f);
}

__global__ void cls2_k(const float* __restrict__ h1c, const float* __restrict__ w2,
                       const float* __restrict__ b2, float* __restrict__ out) {
  int idx = blockIdx.x * blockDim.x + threadIdx.x;
  if (idx >= NB * NUM_CLASSES) return;
  int j = idx % NUM_CLASSES, b = idx / NUM_CLASSES;
  const float* hr = h1c + (size_t)b * HID;
  const float* wr = w2 + (size_t)j * HID;
  float s = b2[j];
  for (int k = 0; k < HID; ++k) s += hr[k] * wr[k];
  out[idx] = s;
}

// =================================================================================
extern "C" void kernel_launch(void* const* d_in, const int* in_sizes, int n_in,
                              void* d_out, int out_size, void* d_ws, size_t ws_size,
                              hipStream_t stream) {
  const float* x        = (const float*)d_in[0];
  const float* mask     = (const float*)d_in[1];
  const float* c1w      = (const float*)d_in[2];
  const float* c1b      = (const float*)d_in[3];
  const float* bn1      = (const float*)d_in[4];
  const float* c2w      = (const float*)d_in[5];
  const float* c2b      = (const float*)d_in[6];
  const float* bn2      = (const float*)d_in[7];
  const float* c3w      = (const float*)d_in[8];
  const float* c3b      = (const float*)d_in[9];
  const float* bn3      = (const float*)d_in[10];
  const float* ln_g     = (const float*)d_in[11];
  const float* ln_b     = (const float*)d_in[12];
  const float* in_w     = (const float*)d_in[13];
  const float* cmw      = (const float*)d_in[14];
  const float* cmb      = (const float*)d_in[15];
  const float* dt_bias  = (const float*)d_in[16];
  const float* A_log    = (const float*)d_in[17];
  const float* Dp       = (const float*)d_in[18];
  const float* rms_w    = (const float*)d_in[19];
  const float* out_w    = (const float*)d_in[20];
  const float* cls_w1   = (const float*)d_in[21];
  const float* cls_b1   = (const float*)d_in[22];
  const float* cls_bn   = (const float*)d_in[23];
  const float* cls_w2   = (const float*)d_in[24];
  const float* cls_b2   = (const float*)d_in[25];
  float* outp = (float*)d_out;

  // -------- workspace carve-out --------
  char* base = (char*)d_ws;
  size_t off = 0;
  auto carve = [&](size_t bytes) -> char* {
    char* p = base + off;
    off += (bytes + 255) & ~(size_t)255;
    return p;
  };
  float*  h1     = (float*)carve((size_t)NB * 192 * 1250 * 4);
  float*  h2     = (float*)carve((size_t)NB * 384 * 625 * 4);
  float*  hs     = (float*)carve((size_t)MROWS * D_MODEL * 4);
  __bf16* Abf    = (__bf16*)carve((size_t)MPAD * D_INNER * 2);      // reused (K=768 / K=1536)
  __bf16* Wbf    = (__bf16*)carve((size_t)NPAD_IN * D_MODEL * 2);   // max weight panel
  float*  zx     = (float*)carve((size_t)MPAD * NPAD_IN * 4);
  float*  xbc    = (float*)carve((size_t)NB * LSEQ * CONV_DIM * 4);
  float*  dtb    = (float*)carve((size_t)NB * LSEQ * NHEADS * 4);
  float*  acs    = (float*)carve((size_t)NB * NCH * NHEADS * CHUNKQ * 4);
  float*  scores = (float*)carve((size_t)NB * NCH * CHUNKQ * CHUNKQ * 4);
  float*  ybuf   = (float*)carve((size_t)NB * LSEQ * NHEADS * HEADDIM * 4);
  float*  states = (float*)carve((size_t)NB * NCH * NHEADS * HEADDIM * D_STATE * 4);
  float*  gout   = (float*)carve((size_t)MPAD * D_MODEL * 4);
  float*  rep    = (float*)carve((size_t)NB * D_MODEL * 4);
  float*  h1c    = (float*)carve((size_t)NB * HID * 4);
  (void)ws_size; (void)n_in; (void)in_sizes; (void)out_size;

  const int T = 256;
  auto blocks = [](long long n, int t) { return (int)((n + t - 1) / t); };

  // -------- conv front-end --------
  conv_front<<<blocks((long long)NB * 192 * 1250, T), T, 0, stream>>>(
      x, c1w, c1b, bn1, h1, NB, IN_CH, L_IN, 192, 1250, 15, 2, 7, 0);
  conv_front<<<blocks((long long)NB * 384 * 625, T), T, 0, stream>>>(
      h1, c2w, c2b, bn2, h2, NB, 192, 1250, 384, 625, 7, 2, 3, 0);
  conv_front<<<blocks((long long)NB * 768 * 625, T), T, 0, stream>>>(
      h2, c3w, c3b, bn3, hs, NB, 384, 625, 768, 625, 3, 1, 1, 1);  // -> (b,l,768)

  // -------- 4 Mamba2 mixer layers --------
  for (int i = 0; i < NLAYERS; ++i) {
    const float* lg   = ln_g + i * D_MODEL;
    const float* lb   = ln_b + i * D_MODEL;
    const float* wip  = in_w + (size_t)i * D_IN * D_MODEL;
    const float* lcw  = cmw + (size_t)i * CONV_DIM * D_CONV;
    const float* lcb  = cmb + (size_t)i * CONV_DIM;
    const float* ldtb = dt_bias + i * NHEADS;
    const float* lAl  = A_log + i * NHEADS;
    const float* lDp  = Dp + i * NHEADS;
    const float* lrw  = rms_w + (size_t)i * D_INNER;
    const float* lwo  = out_w + (size_t)i * D_MODEL * D_INNER;

    // LN -> bf16 A (MPAD x 768), weight -> bf16 (NPAD_IN x 768)
    ln_to_bf16<<<MPAD, 256, 0, stream>>>(hs, lg, lb, Abf, MROWS, D_MODEL);
    conv_w_bf16<<<blocks((long long)NPAD_IN * D_MODEL, T), T, 0, stream>>>(
        wip, Wbf, D_IN, NPAD_IN, D_MODEL);
    // in_proj GEMM: (MPAD x 768) x (NPAD_IN x 768)^T -> zx (ldc = NPAD_IN)
    {
      int mt = MPAD / 64, nt = NPAD_IN / 32;
      gemm_bf16_wmma<<<blocks((long long)mt * nt, 8), 256, 0, stream>>>(
          Abf, Wbf, zx, mt, nt, D_MODEL, D_MODEL, D_MODEL, NPAD_IN);
    }
    // depthwise conv + silu, dt prep, cumsum
    dwconv_silu<<<blocks((long long)NB * LSEQ * CONV_DIM, T), T, 0, stream>>>(
        zx, lcw, lcb, xbc, NB, LSEQ, NPAD_IN);
    dt_prep<<<blocks((long long)NB * LSEQ * NHEADS, T), T, 0, stream>>>(zx, ldtb, dtb, NPAD_IN);
    cumsum_a<<<blocks((long long)NB * NCH * NHEADS, T), T, 0, stream>>>(dtb, lAl, acs);
    // SSD
    scores_k<<<blocks((long long)NB * NCH * CHUNKQ * CHUNKQ, T), T, 0, stream>>>(xbc, scores);
    ydiag_k<<<NB * NCH * NHEADS, 128, 0, stream>>>(scores, acs, xbc, dtb, ybuf);
    states_k<<<blocks((long long)NB * NCH * NHEADS * HEADDIM * D_STATE, T), T, 0, stream>>>(
        xbc, dtb, acs, states);
    scan_k<<<blocks((long long)NB * NHEADS * HEADDIM * D_STATE, T), T, 0, stream>>>(states, acs);
    yoff_k<<<blocks((long long)NB * LSEQ * NHEADS * HEADDIM, T), T, 0, stream>>>(
        xbc, dtb, acs, states, lDp, ybuf);
    // gate + RMSNorm -> bf16 (MPAD x 1536)
    gate_rms_bf16<<<MPAD, 256, 0, stream>>>(ybuf, zx, lrw, Abf, MROWS, NPAD_IN);
    // out_proj GEMM: (MPAD x 1536) x (768 x 1536)^T -> gout (ldc = 768)
    conv_w_bf16<<<blocks((long long)D_MODEL * D_INNER, T), T, 0, stream>>>(
        lwo, Wbf, D_MODEL, D_MODEL, D_INNER);
    {
      int mt = MPAD / 64, nt = D_MODEL / 32;
      gemm_bf16_wmma<<<blocks((long long)mt * nt, 8), 256, 0, stream>>>(
          Abf, Wbf, gout, mt, nt, D_INNER, D_INNER, D_INNER, D_MODEL);
    }
    add_res<<<blocks((long long)MROWS * D_MODEL, T), T, 0, stream>>>(hs, gout);
  }

  // -------- pool + classifier head --------
  pool_k<<<NB, 256, 0, stream>>>(hs, mask, rep);
  cls1_k<<<blocks((long long)NB * HID, T), T, 0, stream>>>(rep, cls_w1, cls_b1, cls_bn, h1c);
  cls2_k<<<blocks((long long)NB * NUM_CLASSES, T), T, 0, stream>>>(h1c, cls_w2, cls_b2, outp);
}
